// DecoderBlock_25881472925815
// MI455X (gfx1250) — compile-verified, matrix operand order checked by reading, not run on hardware
//
#include <hip/hip_runtime.h>
#include <hip/hip_bf16.h>
#include <math.h>

// ---------------- problem constants ----------------
#define N_RES   2048
#define N_ATOM  5
#define DMODEL  256
#define PDIM    256
#define NHEAD   8
#define DHEAD   32
#define KNBR    64
#define NPAIR   (N_RES * KNBR)
#define RBFB    16
#define DMAXV   22.0f
#define SIGMA   10.0f
#define INFK    1e30f
#define NBATCH  2
#define NCHAIN  4

// Use gfx1250 async global->LDS copies for the A tile (ASYNCcnt path).
#define USE_ASYNC_LDS 1

typedef __bf16 bf16;
typedef __attribute__((ext_vector_type(16))) __bf16 v16bf;
typedef __attribute__((ext_vector_type(8)))  __bf16 v8bf;
typedef __attribute__((ext_vector_type(8)))  float  v8f;

enum { EPI_NONE = 0, EPI_BIAS = 1, EPI_GELU = 2, EPI_BIAS_GELU = 3, EPI_RELPOS_LN = 4 };

// ---------------- device helpers ----------------
__device__ inline float geluf(float x) {
  float x3 = x * x * x;
  return 0.5f * x * (1.f + tanhf(0.79788456f * (x + 0.044715f * x3)));
}

__device__ inline float hash_u01(unsigned x) {
  x ^= x >> 17; x *= 0xed5ad4bbu;
  x ^= x >> 11; x *= 0xac4c1b51u;
  x ^= x >> 15; x *= 0x31848babu;
  x ^= x >> 14;
  return ((float)(x >> 8) + 0.5f) * (1.f / 16777216.f);
}

// LDS byte offset of a __shared__ object (generic -> addrspace(3) -> u32).
__device__ inline unsigned lds_offset(const void* p) {
  return (unsigned)(size_t)(__attribute__((address_space(3))) const void*)p;
}

// Load a 16x32 bf16 fragment (A layout per CDNA5 ISA 7.12.2) from an LDS tile.
// For A: base = tile (row-major, stride elems). For B: tile stored transposed
// (col-major: Bt[n*stride + k]) so the same loader yields the B fragment.
__device__ inline v16bf load_frag(const bf16* base, int stride) {
  int lane = threadIdx.x & 31;
  int r = lane & 15;        // M (A) or N (B)
  int kh = lane >> 4;       // K-half select
  const bf16* p = base + r * stride + kh * 8;
  v8bf lo = *(const v8bf*)(p);        // K = kh*8 .. kh*8+7
  v8bf hi = *(const v8bf*)(p + 16);   // K = 16+kh*8 .. 16+kh*8+7
  v16bf o;
#pragma unroll
  for (int e = 0; e < 8; ++e) { o[e] = lo[e]; o[8 + e] = hi[e]; }
  return o;
}

// ---------------- generic WMMA GEMM ----------------
// C[M x BN] = A[M x K](bf16,row-major,lda) * B[K x BN](bf16,row-major,ldb) (+epilogue)
// One block per 16 rows; 256 threads = 8 waves; each wave owns BN/128 col tiles.
template <int BN, int EPI, bool OUTF32, bool OUTBF16>
__global__ __launch_bounds__(256) void wmma_gemm16_kernel(
    const bf16* __restrict__ Am, int lda,
    const bf16* __restrict__ Bm, int ldb, int Kdim,
    const float* __restrict__ bias,
    const int*  __restrict__ rp,
    const float* __restrict__ relpos,   // 66 x BN table (RELPOS_LN only)
    const float* __restrict__ ln_s, const float* __restrict__ ln_b,
    float* __restrict__ Cf, bf16* __restrict__ Cb, int ldc) {
  constexpr int NT = BN / 128;  // col tiles per wave
  __shared__ __align__(16) bf16 As[16 * 32];
  __shared__ __align__(16) bf16 Bs[BN * 40];  // transposed, padded stride 40
  __shared__ float Ep[(EPI == EPI_RELPOS_LN) ? (16 * BN + 64) : 64];

  const int tid = threadIdx.x;
  const int wave = tid >> 5;
  const int row0 = blockIdx.x * 16;
  v8f acc[NT] = {};

  for (int k0 = 0; k0 < Kdim; k0 += 32) {
    __syncthreads();
#if USE_ASYNC_LDS
    // A tile: 16 rows x 32 bf16 = 64 x 16B segments, DMAed straight into LDS
    // via the gfx1250 async global->LDS path (tracked by ASYNCcnt).
    if (tid < 64) {
      int r = tid >> 2, seg = tid & 3;
      const bf16* gsrc = Am + (size_t)(row0 + r) * lda + k0 + seg * 8;
      unsigned ldst = lds_offset(&As[r * 32 + seg * 8]);
      unsigned long long ga = (unsigned long long)(size_t)gsrc;
      asm volatile("global_load_async_to_lds_b128 %0, %1, off"
                   :: "v"(ldst), "v"(ga)
                   : "memory");
    }
#else
    {  // A tile: 16x32, two contiguous bf16 per thread
      int e = tid * 2;
      int r = e >> 5, c = e & 31;
      const bf16* src = Am + (size_t)(row0 + r) * lda + k0 + c;
      As[r * 32 + c] = src[0];
      As[r * 32 + c + 1] = src[1];
    }
#endif
    // B tile: 32 x BN, stored transposed into Bs[n*40 + k]
    for (int e = tid * 8; e < 32 * BN; e += 256 * 8) {
      int kk = e / BN, nn = e % BN;
      const bf16* src = Bm + (size_t)(k0 + kk) * ldb + nn;
#pragma unroll
      for (int q = 0; q < 8; ++q) Bs[(nn + q) * 40 + kk] = src[q];
    }
    if (k0 + 32 < Kdim) __builtin_prefetch(Bm + (size_t)(k0 + 32) * ldb, 0, 1);
#if USE_ASYNC_LDS
    asm volatile("s_wait_asynccnt 0" ::: "memory");
#endif
    __syncthreads();

    v16bf af = load_frag(As, 32);
#pragma unroll
    for (int j = 0; j < NT; ++j) {
      int col0 = (wave * NT + j) * 16;
      v16bf bfr = load_frag(&Bs[col0 * 40], 40);
      acc[j] = __builtin_amdgcn_wmma_f32_16x16x32_bf16(
          false, af, false, bfr, (short)0, acc[j], false, false);
    }
  }
  __syncthreads();

  const int lane = tid & 31;
  const int ncol = lane & 15;
  const int mg = lane >> 4;

  if constexpr (EPI == EPI_RELPOS_LN) {
#pragma unroll
    for (int j = 0; j < NT; ++j) {
      int col = (wave * NT + j) * 16 + ncol;
#pragma unroll
      for (int r = 0; r < 8; ++r) {
        int m = r + mg * 8;
        int rpi = rp[row0 + m];
        Ep[m * BN + col] = acc[j][r] + relpos[(size_t)rpi * BN + col];
      }
    }
    __syncthreads();
    if (tid < 16) {
      float s = 0.f, s2 = 0.f;
      for (int c = 0; c < BN; ++c) { float v = Ep[tid * BN + c]; s += v; s2 += v * v; }
      float mu = s / (float)BN;
      float var = s2 / (float)BN - mu * mu;
      Ep[16 * BN + tid] = mu;
      Ep[16 * BN + 16 + tid] = rsqrtf(var + 1e-5f);
    }
    __syncthreads();
    for (int e = tid; e < 16 * BN; e += 256) {
      int m = e / BN, c = e - m * BN;
      float v = (Ep[e] - Ep[16 * BN + m]) * Ep[16 * BN + 16 + m] * ln_s[c] + ln_b[c];
      size_t o = (size_t)(row0 + m) * ldc + c;
      if (OUTF32) Cf[o] = v;
      if (OUTBF16) Cb[o] = (bf16)v;
    }
  } else {
#pragma unroll
    for (int j = 0; j < NT; ++j) {
      int col = (wave * NT + j) * 16 + ncol;
#pragma unroll
      for (int r = 0; r < 8; ++r) {
        int m = r + mg * 8;
        float v = acc[j][r];
        if constexpr (EPI == EPI_BIAS || EPI == EPI_BIAS_GELU) v += bias[col];
        if constexpr (EPI == EPI_GELU || EPI == EPI_BIAS_GELU) v = geluf(v);
        size_t o = (size_t)(row0 + m) * ldc + col;
        if (OUTF32) Cf[o] = v;
        if (OUTBF16) Cb[o] = (bf16)v;
      }
    }
  }
}

// ---------------- frames ----------------
__global__ void frames_kernel(const float* __restrict__ pos, float* __restrict__ Rm,
                              float* __restrict__ tv, float* __restrict__ localp) {
  int i = blockIdx.x * blockDim.x + threadIdx.x;
  if (i >= N_RES) return;
  float p0[3], p1[3], p2[3];
#pragma unroll
  for (int c = 0; c < 3; ++c) {
    p0[c] = pos[(i * 5 + 0) * 3 + c];
    p1[c] = pos[(i * 5 + 1) * 3 + c];
    p2[c] = pos[(i * 5 + 2) * 3 + c];
  }
  float e1[3], e2[3], e3[3];
  float n1 = 0.f;
#pragma unroll
  for (int c = 0; c < 3; ++c) { e1[c] = p2[c] - p1[c]; n1 += e1[c] * e1[c]; }
  n1 = sqrtf(n1) + 1e-8f;
#pragma unroll
  for (int c = 0; c < 3; ++c) e1[c] /= n1;
  float v2[3], dot = 0.f;
#pragma unroll
  for (int c = 0; c < 3; ++c) { v2[c] = p0[c] - p1[c]; dot += v2[c] * e1[c]; }
  float n2 = 0.f;
#pragma unroll
  for (int c = 0; c < 3; ++c) { e2[c] = v2[c] - dot * e1[c]; n2 += e2[c] * e2[c]; }
  n2 = sqrtf(n2) + 1e-8f;
#pragma unroll
  for (int c = 0; c < 3; ++c) e2[c] /= n2;
  e3[0] = e1[1] * e2[2] - e1[2] * e2[1];
  e3[1] = e1[2] * e2[0] - e1[0] * e2[2];
  e3[2] = e1[0] * e2[1] - e1[1] * e2[0];
  float R[9];
#pragma unroll
  for (int r = 0; r < 3; ++r) { R[r * 3 + 0] = e1[r]; R[r * 3 + 1] = e2[r]; R[r * 3 + 2] = e3[r]; }
#pragma unroll
  for (int c = 0; c < 9; ++c) Rm[i * 9 + c] = R[c];
#pragma unroll
  for (int c = 0; c < 3; ++c) tv[i * 3 + c] = p1[c];
  for (int a = 0; a < N_ATOM; ++a) {
    float rel[3];
#pragma unroll
    for (int r = 0; r < 3; ++r) rel[r] = pos[(i * 5 + a) * 3 + r] - p1[r];
#pragma unroll
    for (int c = 0; c < 3; ++c) {
      float v = 0.f;
#pragma unroll
      for (int r = 0; r < 3; ++r) v += R[r * 3 + c] * rel[r];  // R^T rel
      localp[i * 15 + a * 3 + c] = v;
    }
  }
}

// ---------------- neighbor selection (top-K with hash-gumbel) ----------------
__global__ __launch_bounds__(256) void nbr_kernel(
    const float* __restrict__ pos, const int* __restrict__ resi,
    const int* __restrict__ chain, const int* __restrict__ batch,
    const unsigned char* __restrict__ mask, int* __restrict__ nb) {
  __shared__ float sd[N_RES];
  __shared__ float sk[N_RES];
  __shared__ unsigned char sf[N_RES];
  __shared__ float rv[256];
  __shared__ int ri[256];
  __shared__ float scut;
  const int i = blockIdx.x, t = threadIdx.x;
  const int bi_ = batch[i], ci_ = chain[i], rsi = resi[i];
  const bool mi = mask[i] != 0;
  float cx = pos[(i * 5 + 1) * 3 + 0], cy = pos[(i * 5 + 1) * 3 + 1], cz = pos[(i * 5 + 1) * 3 + 2];
  for (int j = t; j < N_RES; j += 256) {
    bool sb = (batch[j] == bi_);
    bool valid = sb && mi && (mask[j] != 0);
    bool win = (abs(rsi - resi[j]) < 16) && sb && (chain[j] == ci_);
    float dx = pos[(j * 5 + 1) * 3 + 0] - cx;
    float dy = pos[(j * 5 + 1) * 3 + 1] - cy;
    float dz = pos[(j * 5 + 1) * 3 + 2] - cz;
    float d = sqrtf(dx * dx + dy * dy + dz * dz);
    sf[j] = (unsigned char)((valid ? 1 : 0) | (win ? 2 : 0));
    float de = (win || !valid) ? INFK : d;
    sd[j] = de;
    sk[j] = de;
  }
  __syncthreads();
  // 16th smallest distance (spatial cutoff)
  for (int q = 0; q < 16; ++q) {
    float bv = INFK; int bj = N_RES;
    for (int j = t; j < N_RES; j += 256)
      if (sk[j] < bv || (sk[j] == bv && j < bj)) { bv = sk[j]; bj = j; }
    rv[t] = bv; ri[t] = bj;
    for (int s = 128; s > 0; s >>= 1) {
      __syncthreads();
      if (t < s) {
        if (rv[t + s] < rv[t] || (rv[t + s] == rv[t] && ri[t + s] < ri[t])) {
          rv[t] = rv[t + s]; ri[t] = ri[t + s];
        }
      }
    }
    __syncthreads();
    if (t == 0) { scut = rv[0]; if (ri[0] < N_RES) sk[ri[0]] = INFK; }
    __syncthreads();
  }
  const float cutoff = scut;
  for (int j = t; j < N_RES; j += 256) {
    unsigned char f = sf[j];
    bool valid = f & 1, win = f & 2;
    float d = sd[j];
    bool w2 = win || (d < cutoff);
    float key;
    if (!valid) key = INFK;
    else if (w2) key = -10000.f;
    else {
      float u = hash_u01((unsigned)(i * N_RES + j) * 2654435761u + 42u);
      float g = -logf(-logf(u));
      key = g + 3.f * logf(fmaxf(d, 1e-6f));
    }
    sk[j] = key;
  }
  __syncthreads();
  for (int q = 0; q < KNBR; ++q) {
    float bv = INFK; int bj = N_RES;
    for (int j = t; j < N_RES; j += 256)
      if (sk[j] < bv || (sk[j] == bv && j < bj)) { bv = sk[j]; bj = j; }
    rv[t] = bv; ri[t] = bj;
    for (int s = 128; s > 0; s >>= 1) {
      __syncthreads();
      if (t < s) {
        if (rv[t + s] < rv[t] || (rv[t + s] == rv[t] && ri[t + s] < ri[t])) {
          rv[t] = rv[t + s]; ri[t] = ri[t + s];
        }
      }
    }
    __syncthreads();
    if (t == 0) {
      nb[i * KNBR + q] = (rv[0] < 1e29f) ? ri[0] : -1;
      if (ri[0] < N_RES) sk[ri[0]] = INFK;
    }
    __syncthreads();
  }
}

// ---------------- pair raw features (119-dim padded to 128, bf16) ----------------
__global__ void pair_raw_kernel(const float* __restrict__ pos, const float* __restrict__ Rm,
                                const float* __restrict__ tv, const int* __restrict__ resi,
                                const int* __restrict__ chain, const int* __restrict__ batch,
                                const unsigned char* __restrict__ mask, const int* __restrict__ nb,
                                bf16* __restrict__ feat, int* __restrict__ rp,
                                int* __restrict__ pmask) {
  size_t e = (size_t)blockIdx.x * 256 + threadIdx.x;
  if (e >= (size_t)NPAIR) return;
  int i = (int)(e >> 6);
  int j = nb[e];
  bf16* f = feat + e * 128;
  if (j < 0) {
#pragma unroll
    for (int c = 0; c < 128; ++c) f[c] = (bf16)0.f;
    rp[e] = 65; pmask[e] = 0;
    return;
  }
  pmask[e] = (mask[i] && mask[j]) ? 1 : 0;
  bool other = (chain[j] != chain[i]) || (batch[j] != batch[i]);
  int r = resi[j] - resi[i];
  r = min(32, max(-32, r)) + 32;
  rp[e] = other ? 65 : r;
  float Ri[9], Rj[9], ti[3];
#pragma unroll
  for (int c = 0; c < 9; ++c) { Ri[c] = Rm[i * 9 + c]; Rj[c] = Rm[j * 9 + c]; }
#pragma unroll
  for (int c = 0; c < 3; ++c) ti[c] = tv[i * 3 + c];
  const float W = DMAXV / (float)RBFB;
  for (int a = 0; a < N_ATOM; ++a) {
    float rel[3];
    float d2 = 0.f;
#pragma unroll
    for (int c = 0; c < 3; ++c) {
      rel[c] = pos[(j * 5 + a) * 3 + c] - ti[c];
      d2 += rel[c] * rel[c];
    }
    float d = sqrtf(d2);
#pragma unroll
    for (int b = 0; b < RBFB; ++b) {
      float z = (d - ((float)b + 0.5f) * W) / W;
      f[a * RBFB + b] = (bf16)expf(-z * z);
    }
    float lr[3], ln2 = 0.f;
#pragma unroll
    for (int c = 0; c < 3; ++c) {
      float v = 0.f;
#pragma unroll
      for (int rr = 0; rr < 3; ++rr) v += Ri[rr * 3 + c] * rel[rr];  // R^T rel
      lr[c] = v; ln2 += v * v;
    }
    float inv = 1.f / (sqrtf(ln2) + 1e-8f);
#pragma unroll
    for (int c = 0; c < 3; ++c) {
      f[80 + a * 3 + c] = (bf16)(lr[c] * inv);
      f[104 + a * 3 + c] = (bf16)(lr[c] * (1.f / DMAXV));
    }
  }
  // relR = Ri^T Rj
#pragma unroll
  for (int c = 0; c < 3; ++c)
#pragma unroll
    for (int l = 0; l < 3; ++l) {
      float v = 0.f;
#pragma unroll
      for (int rr = 0; rr < 3; ++rr) v += Ri[rr * 3 + c] * Rj[rr * 3 + l];
      f[95 + c * 3 + l] = (bf16)v;
    }
#pragma unroll
  for (int c = 119; c < 128; ++c) f[c] = (bf16)0.f;
}

// ---------------- layernorm (row of 256) ----------------
__global__ __launch_bounds__(256) void ln_kernel(const float* __restrict__ x,
                                                 const float* __restrict__ s,
                                                 const float* __restrict__ b,
                                                 float* of, bf16* ob) {
  __shared__ float red[256];
  int row = blockIdx.x, t = threadIdx.x;
  float v = x[(size_t)row * DMODEL + t];
  red[t] = v;
  __syncthreads();
  for (int st = 128; st > 0; st >>= 1) {
    if (t < st) red[t] += red[t + st];
    __syncthreads();
  }
  float mu = red[0] * (1.f / DMODEL);
  __syncthreads();
  float d = v - mu;
  red[t] = d * d;
  __syncthreads();
  for (int st = 128; st > 0; st >>= 1) {
    if (t < st) red[t] += red[t + st];
    __syncthreads();
  }
  float var = red[0] * (1.f / DMODEL);
  float y = d * rsqrtf(var + 1e-5f) * s[t] + b[t];
  if (of) of[(size_t)row * DMODEL + t] = y;
  if (ob) ob[(size_t)row * DMODEL + t] = (bf16)y;
}

// ---------------- fused attention: logits + softmax + context ----------------
__global__ __launch_bounds__(256) void attn_kernel(
    const float* __restrict__ q, const float* __restrict__ kmat, const float* __restrict__ vmat,
    const bf16* __restrict__ pair, const int* __restrict__ nb, const int* __restrict__ pmask,
    const float* __restrict__ pos, const float* __restrict__ Wb,
    const float* __restrict__ wdist, bf16* __restrict__ cat) {
  __shared__ float sq[DMODEL];
  __shared__ float slog[KNBR * NHEAD];
  __shared__ float ssp[NHEAD];
  __shared__ float sca[3];
  const int i = blockIdx.x, t = threadIdx.x;
  sq[t] = q[(size_t)i * DMODEL + t];
  if (t < NHEAD) ssp[t] = log1pf(expf(wdist[t]));
  if (t < 3) sca[t] = pos[(i * 5 + 1) * 3 + t] * (1.f / SIGMA);
  __syncthreads();
  for (int it = t; it < KNBR * NHEAD; it += 256) {
    int k_ = it >> 3, h = it & 7;
    int j = nb[i * KNBR + k_];
    float lg = -1e9f;
    if (j >= 0 && pmask[i * KNBR + k_]) {
      float dot = 0.f;
#pragma unroll
      for (int d = 0; d < DHEAD; ++d) dot += sq[h * DHEAD + d] * kmat[(size_t)j * DMODEL + h * DHEAD + d];
      dot *= 0.1767766953f;  // 1/sqrt(32)
      float pl = 0.f;
      const bf16* pr = pair + ((size_t)i * KNBR + k_) * PDIM;
      for (int p = 0; p < PDIM; ++p) pl += (float)pr[p] * Wb[p * NHEAD + h];
      float dx = pos[(j * 5 + 1) * 3 + 0] * (1.f / SIGMA) - sca[0];
      float dy = pos[(j * 5 + 1) * 3 + 1] * (1.f / SIGMA) - sca[1];
      float dz = pos[(j * 5 + 1) * 3 + 2] * (1.f / SIGMA) - sca[2];
      float dist = sqrtf(dx * dx + dy * dy + dz * dz);
      lg = dot + pl - ssp[h] * dist;
    }
    slog[k_ * NHEAD + h] = lg;
  }
  __syncthreads();
  if (t < NHEAD) {
    float mx = -INFK;
    for (int k_ = 0; k_ < KNBR; ++k_) mx = fmaxf(mx, slog[k_ * NHEAD + t]);
    float sm = 0.f;
    for (int k_ = 0; k_ < KNBR; ++k_) {
      float ev = expf(slog[k_ * NHEAD + t] - mx);
      slog[k_ * NHEAD + t] = ev;
      sm += ev;
    }
    float inv = 1.f / sm;
    for (int k_ = 0; k_ < KNBR; ++k_) slog[k_ * NHEAD + t] *= inv;
  }
  __syncthreads();
  {  // out = sum_k attn * v[nb]
    int h = t >> 5;
    float o = 0.f;
    for (int k_ = 0; k_ < KNBR; ++k_) {
      int j = nb[i * KNBR + k_];
      if (j >= 0) o += slog[k_ * NHEAD + h] * vmat[(size_t)j * DMODEL + t];
    }
    cat[(size_t)i * 2304 + t] = (bf16)o;
  }
  for (int e = t; e < NHEAD * PDIM; e += 256) {  // pout = sum_k attn * pair
    int h = e >> 8, p = e & 255;
    float o = 0.f;
    const bf16* pr = pair + (size_t)i * KNBR * PDIM + p;
    for (int k_ = 0; k_ < KNBR; ++k_) o += slog[k_ * NHEAD + h] * (float)pr[(size_t)k_ * PDIM];
    cat[(size_t)i * 2304 + DMODEL + e] = (bf16)o;
  }
}

// ---------------- small utility kernels ----------------
__global__ void conv_kernel(const float* __restrict__ s, bf16* __restrict__ d, int n) {
  int i = blockIdx.x * 256 + threadIdx.x;
  if (i < n) d[i] = (bf16)s[i];
}
__global__ void zero_kernel(float* p, int n) {
  int i = blockIdx.x * 256 + threadIdx.x;
  if (i < n) p[i] = 0.f;
}
__global__ void add_kernel(const float* a, const float* b, float* c, int n) {
  int i = blockIdx.x * 256 + threadIdx.x;
  if (i < n) c[i] = a[i] + b[i];
}
__global__ void addconv_kernel(const float* a, const float* b, float* cf, bf16* cb, int n) {
  int i = blockIdx.x * 256 + threadIdx.x;
  if (i < n) { float v = a[i] + b[i]; cf[i] = v; cb[i] = (bf16)v; }
}
__global__ void padrows_kernel(const float* __restrict__ src, bf16* __restrict__ dst,
                               int rows, int cols, int rows_pad) {
  int e = blockIdx.x * 256 + threadIdx.x;
  if (e >= rows_pad * cols) return;
  int r = e / cols, c = e - r * cols;
  dst[e] = (r < rows) ? (bf16)src[r * cols + c] : (bf16)0.f;
}
__global__ void wcat_kernel(const float* __restrict__ Wdist, const float* __restrict__ Wdir,
                            const float* __restrict__ Wrot, const float* __restrict__ Wpvec,
                            bf16* __restrict__ dst) {
  int e = blockIdx.x * 256 + threadIdx.x;
  if (e >= 128 * PDIM) return;
  int r = e >> 8, c = e & 255;
  float v = 0.f;
  if (r < 80) v = Wdist[r * PDIM + c];
  else if (r < 95) v = Wdir[(r - 80) * PDIM + c];
  else if (r < 104) v = Wrot[(r - 95) * PDIM + c];
  else if (r < 119) v = Wpvec[(r - 104) * PDIM + c];
  dst[e] = (bf16)v;
}
__global__ void lppad_kernel(const float* __restrict__ localp, bf16* __restrict__ lpb) {
  int e = blockIdx.x * 256 + threadIdx.x;
  if (e >= N_RES * 32) return;
  int i = e >> 5, c = e & 31;
  lpb[e] = (c < 15) ? (bf16)localp[i * 15 + c] : (bf16)0.f;
}

// segment-mean accumulation + gather
__global__ void seg_accum_kernel(const float* __restrict__ lu, const float* __restrict__ bg,
                                 const float* __restrict__ cg, const int* __restrict__ batch,
                                 const int* __restrict__ chain, const unsigned char* __restrict__ mask,
                                 float* segB, float* wB, float* segC, float* wC) {
  int e = blockIdx.x * 256 + threadIdx.x;
  if (e >= N_RES * 1024) return;
  int i = e >> 10, c = e & 1023;
  float w = mask[i] ? 1.f : 0.f;
  float l = lu[e];
  atomicAdd(&segB[batch[i] * 1024 + c], bg[e] * l * w);
  atomicAdd(&segC[chain[i] * 1024 + c], cg[e] * l * w);
  if (c == 0) { atomicAdd(&wB[batch[i]], w); atomicAdd(&wC[chain[i]], w); }
}
__global__ void hidden_kernel(const float* __restrict__ lu, const float* __restrict__ lg,
                              const float* __restrict__ segB, const float* __restrict__ wB,
                              const float* __restrict__ segC, const float* __restrict__ wC,
                              const int* __restrict__ batch, const int* __restrict__ chain,
                              bf16* __restrict__ hid) {
  int e = blockIdx.x * 256 + threadIdx.x;
  if (e >= N_RES * 1024) return;
  int i = e >> 10, c = e & 1023;
  float h = segB[batch[i] * 1024 + c] / fmaxf(wB[batch[i]], 1e-6f) +
            segC[chain[i] * 1024 + c] / fmaxf(wC[chain[i]], 1e-6f) + lg[e] * lu[e];
  hid[e] = (bf16)h;
}

// ---------------- position head ----------------
__global__ void posdelta_kernel(const float* __restrict__ ln3, const float* __restrict__ Wpos,
                                float* __restrict__ pd) {
  int e = blockIdx.x * 256 + threadIdx.x;
  if (e >= N_RES * 15) return;
  int i = e / 15, m = e - i * 15;
  float v = 0.f;
  for (int d = 0; d < DMODEL; ++d) v += ln3[(size_t)i * DMODEL + d] * Wpos[d * 15 + m];
  pd[e] = v;
}
__global__ void posout_kernel(const float* __restrict__ localp, const float* __restrict__ pd,
                              const float* __restrict__ Rm, const float* __restrict__ tv,
                              float* __restrict__ outpos, float* __restrict__ outzero) {
  int i = blockIdx.x * blockDim.x + threadIdx.x;
  if (i == 0) outzero[0] = 0.f;
  if (i >= N_RES) return;
  float R[9];
#pragma unroll
  for (int c = 0; c < 9; ++c) R[c] = Rm[i * 9 + c];
  for (int a = 0; a < N_ATOM; ++a) {
    float lp2[3];
#pragma unroll
    for (int c = 0; c < 3; ++c) lp2[c] = localp[i * 15 + a * 3 + c] + SIGMA * pd[i * 15 + a * 3 + c];
#pragma unroll
    for (int c = 0; c < 3; ++c) {
      float v = 0.f;
#pragma unroll
      for (int jj = 0; jj < 3; ++jj) v += R[c * 3 + jj] * lp2[jj];
      outpos[(i * 5 + a) * 3 + c] = v + tv[i * 3 + c];
    }
  }
}

// ---------------- host ----------------
static inline int cdiv(int a, int b) { return (a + b - 1) / b; }

extern "C" void kernel_launch(void* const* d_in, const int* in_sizes, int n_in,
                              void* d_out, int out_size, void* d_ws, size_t ws_size,
                              hipStream_t stream) {
  (void)in_sizes; (void)n_in; (void)out_size; (void)ws_size;
  // ---- inputs ----
  const float* features = (const float*)d_in[0];
  const float* pos      = (const float*)d_in[1];
  const int*   resi     = (const int*)d_in[2];
  const int*   chain    = (const int*)d_in[3];
  const int*   batch    = (const int*)d_in[4];
  const unsigned char* mask = (const unsigned char*)d_in[5];
  // ---- params (dict order) ----
  const float* W_relpos = (const float*)d_in[6];
  const float* W_dist   = (const float*)d_in[7];
  const float* W_dir    = (const float*)d_in[8];
  const float* W_rot    = (const float*)d_in[9];
  const float* W_pvec   = (const float*)d_in[10];
  const float* ln_pair_s = (const float*)d_in[11];
  const float* ln_pair_b = (const float*)d_in[12];
  const float* W_pmlp1  = (const float*)d_in[13];
  const float* b_pmlp1  = (const float*)d_in[14];
  const float* W_pmlp2  = (const float*)d_in[15];
  const float* b_pmlp2  = (const float*)d_in[16];
  const float* ln1_s = (const float*)d_in[17];
  const float* ln1_b = (const float*)d_in[18];
  const float* Wq = (const float*)d_in[19];
  const float* Wk = (const float*)d_in[20];
  const float* Wv = (const float*)d_in[21];
  const float* Wb = (const float*)d_in[22];
  const float* w_dist_attn = (const float*)d_in[23];
  const float* Wo = (const float*)d_in[24];
  const float* bo = (const float*)d_in[25];
  const float* ln2_s = (const float*)d_in[26];
  const float* ln2_b = (const float*)d_in[27];
  const float* W_u1 = (const float*)d_in[28];
  const float* b_u1 = (const float*)d_in[29];
  const float* W_u2 = (const float*)d_in[30];
  const float* b_u2 = (const float*)d_in[31];
  const float* W_lu = (const float*)d_in[32];
  const float* W_lg = (const float*)d_in[33];
  const float* W_cg = (const float*)d_in[34];
  const float* W_bg = (const float*)d_in[35];
  const float* W_upd = (const float*)d_in[36];
  const float* b_upd = (const float*)d_in[37];
  const float* ln3_s = (const float*)d_in[38];
  const float* ln3_b = (const float*)d_in[39];
  const float* W_pos = (const float*)d_in[40];

  // ---- workspace carve-up ----
  char* wsp = (char*)d_ws;
  size_t off = 0;
  auto alloc = [&](size_t bytes) -> void* {
    void* p = wsp + off;
    off = (off + bytes + 255) & ~(size_t)255;
    return p;
  };
  float* Rb     = (float*)alloc((size_t)N_RES * 9 * 4);
  float* tb     = (float*)alloc((size_t)N_RES * 3 * 4);
  float* lposb  = (float*)alloc((size_t)N_RES * 15 * 4);
  int*   nbb    = (int*)alloc((size_t)NPAIR * 4);
  int*   rpb    = (int*)alloc((size_t)NPAIR * 4);
  int*   pmb    = (int*)alloc((size_t)NPAIR * 4);
  bf16* featraw = (bf16*)alloc((size_t)NPAIR * 128 * 2);
  bf16* pairln  = (bf16*)alloc((size_t)NPAIR * 256 * 2);
  bf16* pairh   = (bf16*)alloc((size_t)NPAIR * 512 * 2);
  bf16* pairf   = (bf16*)alloc((size_t)NPAIR * 256 * 2);
  bf16* xlnb    = (bf16*)alloc((size_t)N_RES * 256 * 2);
  float* qb = (float*)alloc((size_t)N_RES * 256 * 4);
  float* kb = (float*)alloc((size_t)N_RES * 256 * 4);
  float* vb = (float*)alloc((size_t)N_RES * 256 * 4);
  bf16* catb    = (bf16*)alloc((size_t)N_RES * 2304 * 2);
  float* attnout = (float*)alloc((size_t)N_RES * 256 * 4);
  float* feat1   = (float*)alloc((size_t)N_RES * 256 * 4);
  float* local0  = (float*)alloc((size_t)N_RES * 256 * 4);
  bf16* lpb      = (bf16*)alloc((size_t)N_RES * 32 * 2);
  bf16* hub      = (bf16*)alloc((size_t)N_RES * 512 * 2);
  float* uout    = (float*)alloc((size_t)N_RES * 256 * 4);
  float* localf  = (float*)alloc((size_t)N_RES * 256 * 4);
  bf16* localb   = (bf16*)alloc((size_t)N_RES * 256 * 2);
  float* lub = (float*)alloc((size_t)N_RES * 1024 * 4);
  float* lgb = (float*)alloc((size_t)N_RES * 1024 * 4);
  float* cgb = (float*)alloc((size_t)N_RES * 1024 * 4);
  float* bgb = (float*)alloc((size_t)N_RES * 1024 * 4);
  float* segB = (float*)alloc((size_t)NBATCH * 1024 * 4);
  float* wBv  = (float*)alloc((size_t)NBATCH * 4);
  float* segC = (float*)alloc((size_t)NCHAIN * 1024 * 4);
  float* wCv  = (float*)alloc((size_t)NCHAIN * 4);
  bf16* hidb  = (bf16*)alloc((size_t)N_RES * 1024 * 2);
  float* updout = (float*)alloc((size_t)N_RES * 256 * 4);
  float* ln3f   = (float*)alloc((size_t)N_RES * 256 * 4);
  float* pdel   = (float*)alloc((size_t)N_RES * 15 * 4);
  // bf16 weights
  bf16* Wcatb = (bf16*)alloc((size_t)128 * 256 * 2);
  bf16* W1b   = (bf16*)alloc((size_t)256 * 512 * 2);
  bf16* W2b   = (bf16*)alloc((size_t)512 * 256 * 2);
  bf16* Wqb   = (bf16*)alloc((size_t)256 * 256 * 2);
  bf16* Wkb   = (bf16*)alloc((size_t)256 * 256 * 2);
  bf16* Wvb   = (bf16*)alloc((size_t)256 * 256 * 2);
  bf16* Wob   = (bf16*)alloc((size_t)2304 * 256 * 2);
  bf16* Wu1b  = (bf16*)alloc((size_t)32 * 512 * 2);
  bf16* Wu2b  = (bf16*)alloc((size_t)512 * 256 * 2);
  bf16* Wlub  = (bf16*)alloc((size_t)256 * 1024 * 2);
  bf16* Wlgb  = (bf16*)alloc((size_t)256 * 1024 * 2);
  bf16* Wcgb  = (bf16*)alloc((size_t)256 * 1024 * 2);
  bf16* Wbgb  = (bf16*)alloc((size_t)256 * 1024 * 2);
  bf16* Wupdb = (bf16*)alloc((size_t)1024 * 256 * 2);

  float* out_features = (float*)d_out;                      // N*D
  float* out_pos = out_features + (size_t)N_RES * DMODEL;   // N*A*3
  float* out_zero = out_pos + (size_t)N_RES * N_ATOM * 3;   // 1

  // ---- weight conversion ----
  conv_kernel<<<cdiv(256 * 512, 256), 256, 0, stream>>>(W_pmlp1, W1b, 256 * 512);
  conv_kernel<<<cdiv(512 * 256, 256), 256, 0, stream>>>(W_pmlp2, W2b, 512 * 256);
  conv_kernel<<<cdiv(256 * 256, 256), 256, 0, stream>>>(Wq, Wqb, 256 * 256);
  conv_kernel<<<cdiv(256 * 256, 256), 256, 0, stream>>>(Wk, Wkb, 256 * 256);
  conv_kernel<<<cdiv(256 * 256, 256), 256, 0, stream>>>(Wv, Wvb, 256 * 256);
  conv_kernel<<<cdiv(2304 * 256, 256), 256, 0, stream>>>(Wo, Wob, 2304 * 256);
  conv_kernel<<<cdiv(512 * 256, 256), 256, 0, stream>>>(W_u2, Wu2b, 512 * 256);
  conv_kernel<<<cdiv(256 * 1024, 256), 256, 0, stream>>>(W_lu, Wlub, 256 * 1024);
  conv_kernel<<<cdiv(256 * 1024, 256), 256, 0, stream>>>(W_lg, Wlgb, 256 * 1024);
  conv_kernel<<<cdiv(256 * 1024, 256), 256, 0, stream>>>(W_cg, Wcgb, 256 * 1024);
  conv_kernel<<<cdiv(256 * 1024, 256), 256, 0, stream>>>(W_bg, Wbgb, 256 * 1024);
  conv_kernel<<<cdiv(1024 * 256, 256), 256, 0, stream>>>(W_upd, Wupdb, 1024 * 256);
  padrows_kernel<<<cdiv(32 * 512, 256), 256, 0, stream>>>(W_u1, Wu1b, 15, 512, 32);
  wcat_kernel<<<cdiv(128 * 256, 256), 256, 0, stream>>>(W_dist, W_dir, W_rot, W_pvec, Wcatb);

  // ---- geometry + neighbors + pair raw features ----
  frames_kernel<<<cdiv(N_RES, 64), 64, 0, stream>>>(pos, Rb, tb, lposb);
  nbr_kernel<<<N_RES, 256, 0, stream>>>(pos, resi, chain, batch, mask, nbb);
  pair_raw_kernel<<<cdiv(NPAIR, 256), 256, 0, stream>>>(pos, Rb, tb, resi, chain, batch, mask,
                                                        nbb, featraw, rpb, pmb);

  // ---- pair MLP (WMMA) ----
  wmma_gemm16_kernel<256, EPI_RELPOS_LN, false, true><<<NPAIR / 16, 256, 0, stream>>>(
      featraw, 128, Wcatb, 256, 128, nullptr, rpb, W_relpos, ln_pair_s, ln_pair_b,
      nullptr, pairln, 256);
  wmma_gemm16_kernel<512, EPI_BIAS_GELU, false, true><<<NPAIR / 16, 256, 0, stream>>>(
      pairln, 256, W1b, 512, 256, b_pmlp1, nullptr, nullptr, nullptr, nullptr,
      nullptr, pairh, 512);
  wmma_gemm16_kernel<256, EPI_BIAS, false, true><<<NPAIR / 16, 256, 0, stream>>>(
      pairh, 512, W2b, 256, 512, b_pmlp2, nullptr, nullptr, nullptr, nullptr,
      nullptr, pairf, 256);

  // ---- attention ----
  ln_kernel<<<N_RES, 256, 0, stream>>>(features, ln1_s, ln1_b, nullptr, xlnb);
  wmma_gemm16_kernel<256, EPI_NONE, true, false><<<N_RES / 16, 256, 0, stream>>>(
      xlnb, 256, Wqb, 256, 256, nullptr, nullptr, nullptr, nullptr, nullptr, qb, nullptr, 256);
  wmma_gemm16_kernel<256, EPI_NONE, true, false><<<N_RES / 16, 256, 0, stream>>>(
      xlnb, 256, Wkb, 256, 256, nullptr, nullptr, nullptr, nullptr, nullptr, kb, nullptr, 256);
  wmma_gemm16_kernel<256, EPI_NONE, true, false><<<N_RES / 16, 256, 0, stream>>>(
      xlnb, 256, Wvb, 256, 256, nullptr, nullptr, nullptr, nullptr, nullptr, vb, nullptr, 256);
  attn_kernel<<<N_RES, 256, 0, stream>>>(qb, kb, vb, pairf, nbb, pmb, pos, Wb, w_dist_attn, catb);
  wmma_gemm16_kernel<256, EPI_BIAS, true, false><<<N_RES / 16, 256, 0, stream>>>(
      catb, 2304, Wob, 256, 2304, bo, nullptr, nullptr, nullptr, nullptr, attnout, nullptr, 256);
  add_kernel<<<cdiv(N_RES * 256, 256), 256, 0, stream>>>(features, attnout, feat1, N_RES * 256);

  // ---- local update ----
  ln_kernel<<<N_RES, 256, 0, stream>>>(feat1, ln2_s, ln2_b, local0, nullptr);
  lppad_kernel<<<cdiv(N_RES * 32, 256), 256, 0, stream>>>(lposb, lpb);
  wmma_gemm16_kernel<512, EPI_BIAS_GELU, false, true><<<N_RES / 16, 256, 0, stream>>>(
      lpb, 32, Wu1b, 512, 32, b_u1, nullptr, nullptr, nullptr, nullptr, nullptr, hub, 512);
  wmma_gemm16_kernel<256, EPI_BIAS, true, false><<<N_RES / 16, 256, 0, stream>>>(
      hub, 512, Wu2b, 256, 512, b_u2, nullptr, nullptr, nullptr, nullptr, uout, nullptr, 256);
  addconv_kernel<<<cdiv(N_RES * 256, 256), 256, 0, stream>>>(local0, uout, localf, localb,
                                                             N_RES * 256);
  // gates: 4 weights of 256x1024, split into two 512-col chunks each
  for (int half = 0; half < 2; ++half) {
    int co = half * 512;
    wmma_gemm16_kernel<512, EPI_NONE, true, false><<<N_RES / 16, 256, 0, stream>>>(
        localb, 256, Wlub + co, 1024, 256, nullptr, nullptr, nullptr, nullptr, nullptr,
        lub + co, nullptr, 1024);
    wmma_gemm16_kernel<512, EPI_GELU, true, false><<<N_RES / 16, 256, 0, stream>>>(
        localb, 256, Wlgb + co, 1024, 256, nullptr, nullptr, nullptr, nullptr, nullptr,
        lgb + co, nullptr, 1024);
    wmma_gemm16_kernel<512, EPI_GELU, true, false><<<N_RES / 16, 256, 0, stream>>>(
        localb, 256, Wcgb + co, 1024, 256, nullptr, nullptr, nullptr, nullptr, nullptr,
        cgb + co, nullptr, 1024);
    wmma_gemm16_kernel<512, EPI_GELU, true, false><<<N_RES / 16, 256, 0, stream>>>(
        localb, 256, Wbgb + co, 1024, 256, nullptr, nullptr, nullptr, nullptr, nullptr,
        bgb + co, nullptr, 1024);
  }
  // segment means
  zero_kernel<<<cdiv(NBATCH * 1024, 256), 256, 0, stream>>>(segB, NBATCH * 1024);
  zero_kernel<<<1, 256, 0, stream>>>(wBv, NBATCH);
  zero_kernel<<<cdiv(NCHAIN * 1024, 256), 256, 0, stream>>>(segC, NCHAIN * 1024);
  zero_kernel<<<1, 256, 0, stream>>>(wCv, NCHAIN);
  seg_accum_kernel<<<cdiv(N_RES * 1024, 256), 256, 0, stream>>>(lub, bgb, cgb, batch, chain, mask,
                                                                segB, wBv, segC, wCv);
  hidden_kernel<<<cdiv(N_RES * 1024, 256), 256, 0, stream>>>(lub, lgb, segB, wBv, segC, wCv,
                                                             batch, chain, hidb);
  wmma_gemm16_kernel<256, EPI_BIAS, true, false><<<N_RES / 16, 256, 0, stream>>>(
      hidb, 1024, Wupdb, 256, 1024, b_upd, nullptr, nullptr, nullptr, nullptr,
      updout, nullptr, 256);
  add_kernel<<<cdiv(N_RES * 256, 256), 256, 0, stream>>>(feat1, updout, out_features, N_RES * 256);

  // ---- position head ----
  ln_kernel<<<N_RES, 256, 0, stream>>>(out_features, ln3_s, ln3_b, ln3f, nullptr);
  posdelta_kernel<<<cdiv(N_RES * 15, 256), 256, 0, stream>>>(ln3f, W_pos, pdel);
  posout_kernel<<<cdiv(N_RES, 64), 64, 0, stream>>>(lposb, pdel, Rb, tb, out_pos, out_zero);
}
